// GraFormer_74620761801441
// MI455X (gfx1250) — compile-verified, matrix operand order checked by reading, not run on hardware
//
#include <hip/hip_runtime.h>
#include <hip/hip_bf16.h>

typedef __attribute__((ext_vector_type(16))) _Float16 v16h;
typedef __attribute__((ext_vector_type(8)))  _Float16 v8h;
typedef __attribute__((ext_vector_type(2)))  _Float16 h2;
typedef __attribute__((ext_vector_type(8)))  float    v8f;

#define NPTS 21
#define HID 128
#define NLAYERS 4
#define NHEAD 4
#define BATCH 8192
#define BT 6
#define ROWS (BT * NPTS)   // 126
#define RP 128             // padded rows
#define TPB 256
#define NW 8
#define LDH 256            // LDS row stride (halves) for f16 staging buffers

// per-layer half offsets inside weight workspace (f16, [N][K] transposed)
#define OFF_WQ   0
#define OFF_WK   16384
#define OFF_WV   32768
#define OFF_WO   49152
#define OFF_WG1  65536     // 256n x 128k
#define OFF_WG2  98304     // 128n x 256k
#define OFF_WC1  131072    // 3 x (128n x 128k)
#define OFF_WC2  180224
#define LAYER_HALVES 229376
#define TOTAL_HALVES (4 * LAYER_HALVES)

// ---------------------------------------------------------------- prep kernel
__global__ void graformer_prep(const float* __restrict__ Wq,
                               const float* __restrict__ Wk,
                               const float* __restrict__ Wv,
                               const float* __restrict__ Wo,
                               const float* __restrict__ Wg1,
                               const float* __restrict__ Wg2,
                               const float* __restrict__ Wc1,
                               const float* __restrict__ Wc2,
                               _Float16* __restrict__ ws) {
  for (int e = blockIdx.x * blockDim.x + threadIdx.x; e < TOTAL_HALVES;
       e += gridDim.x * blockDim.x) {
    int l = e / LAYER_HALVES;
    int r = e - l * LAYER_HALVES;
    float v;
    if (r < OFF_WK) {
      int n = r >> 7, k = r & 127;
      v = Wq[l * 16384 + k * 128 + n];
    } else if (r < OFF_WV) {
      int rr = r - OFF_WK; int n = rr >> 7, k = rr & 127;
      v = Wk[l * 16384 + k * 128 + n];
    } else if (r < OFF_WO) {
      int rr = r - OFF_WV; int n = rr >> 7, k = rr & 127;
      v = Wv[l * 16384 + k * 128 + n];
    } else if (r < OFF_WG1) {
      int rr = r - OFF_WO; int n = rr >> 7, k = rr & 127;
      v = Wo[l * 16384 + k * 128 + n];
    } else if (r < OFF_WG2) {               // Wg1T: n<256, k<128
      int rr = r - OFF_WG1; int n = rr >> 7, k = rr & 127;
      v = Wg1[l * 32768 + k * 256 + n];
    } else if (r < OFF_WC1) {               // Wg2T: n<128, k<256
      int rr = r - OFF_WG2; int n = rr >> 8, k = rr & 255;
      v = Wg2[l * 32768 + k * 128 + n];
    } else if (r < OFF_WC2) {               // Wc1T: [3][128n][128k]
      int rr = r - OFF_WC1; int p = rr >> 14; int q = rr & 16383;
      int n = q >> 7, k = q & 127;
      v = Wc1[l * 49152 + p * 16384 + k * 128 + n];
    } else {                                // Wc2T
      int rr = r - OFF_WC2; int p = rr >> 14; int q = rr & 16383;
      int n = q >> 7, k = q & 127;
      v = Wc2[l * 49152 + p * 16384 + k * 128 + n];
    }
    ws[e] = (_Float16)v;
  }
}

// ---------------------------------------------------------------- WMMA helpers
// 16-bit operand fragment (ISA 7.12.2): lane<16 -> K {0..7,16..23},
// lane>=16 -> K {8..15,24..31}; major index (M for A / N for B) = lane&15.
// Two contiguous 16-byte runs -> two b128 loads.
__device__ inline v16h ldfrag(const _Float16* p, int ld, int major0, int k0,
                              int lane) {
  const _Float16* q = p + (major0 + (lane & 15)) * ld + k0 + ((lane & 16) ? 8 : 0);
  v8h lo = *(const v8h*)q;
  v8h hi = *(const v8h*)(q + 16);
  return __builtin_shufflevector(lo, hi, 0, 1, 2, 3, 4, 5, 6, 7, 8, 9, 10, 11,
                                 12, 13, 14, 15);
}

__device__ inline v8f mm_part(v8f c, const _Float16* A, int ldA, int m0,
                              const _Float16* WT, int K, int n0, int lane) {
#pragma unroll
  for (int k0 = 0; k0 < 256; k0 += 32) {
    if (k0 >= K) break;
    v16h a = ldfrag(A, ldA, m0, k0, lane);
    v16h b = ldfrag(WT, K, n0, k0, lane);
    c = __builtin_amdgcn_wmma_f32_16x16x32_f16(false, a, false, b, (short)0, c,
                                               false, false);
  }
  return c;
}

// ---------------------------------------------------------------- shared state
struct SMem {
  alignas(16) float    actF[RP * HID];   // fp32 residual stream
  alignas(16) _Float16 sA[RP * LDH];     // staging A (h | V | g ...)
  alignas(16) _Float16 sB[RP * LDH];     // staging B (Q|K, t1|t2, mixed ...)
  float    lm[NPTS * NPTS];
  float    lg[NPTS * NPTS];
  float    dcoef[NPTS];
  float    xl[ROWS * 2];
  float    t1c[ROWS * 2];
  float    t2c[ROWS * 2];
  unsigned char maskl[NPTS];
};

// ---------------------------------------------------------------- device helpers
__device__ inline void mix_nodes(const float* Lmat, const _Float16* src, int sld,
                                 _Float16* dst, int dld, int width,
                                 const _Float16* sub, int subld) {
  // dst = L@src   (or 2*L@src - sub when sub != null); 2 channels per thread
  const int w2 = width >> 1;
  for (int idx = threadIdx.x; idx < ROWS * w2; idx += TPB) {
    int row = idx / w2, c = (idx - row * w2) * 2;
    int b = row / NPTS, n = row - b * NPTS, base = b * NPTS;
    float a0 = 0.f, a1 = 0.f;
#pragma unroll
    for (int j = 0; j < NPTS; ++j) {
      float l = Lmat[n * NPTS + j];
      h2 v = *(const h2*)(src + (base + j) * sld + c);
      a0 += l * (float)v[0];
      a1 += l * (float)v[1];
    }
    if (sub) {
      h2 s = *(const h2*)(sub + row * subld + c);
      a0 = 2.f * a0 - (float)s[0];
      a1 = 2.f * a1 - (float)s[1];
    }
    h2 o;
    o[0] = (_Float16)a0;
    o[1] = (_Float16)a1;
    *(h2*)(dst + row * dld + c) = o;
  }
}

__device__ inline void layer_norm(const float* actF, const float* a,
                                  const float* bv, _Float16* dst, int wid,
                                  int lane) {
  for (int row = wid; row < ROWS; row += NW) {
    float x0 = actF[row * HID + lane];
    float x1 = actF[row * HID + lane + 32];
    float x2 = actF[row * HID + lane + 64];
    float x3 = actF[row * HID + lane + 96];
    float s = x0 + x1 + x2 + x3;
#pragma unroll
    for (int off = 16; off > 0; off >>= 1) s += __shfl_xor(s, off, 32);
    float mean = s * (1.f / HID);
    float d0 = x0 - mean, d1 = x1 - mean, d2 = x2 - mean, d3 = x3 - mean;
    float ss = d0 * d0 + d1 * d1 + d2 * d2 + d3 * d3;
#pragma unroll
    for (int off = 16; off > 0; off >>= 1) ss += __shfl_xor(ss, off, 32);
    float inv = 1.f / (sqrtf(ss * (1.f / (HID - 1))) + 1e-6f);
    dst[row * LDH + lane]      = (_Float16)(a[lane] * d0 * inv + bv[lane]);
    dst[row * LDH + lane + 32] = (_Float16)(a[lane + 32] * d1 * inv + bv[lane + 32]);
    dst[row * LDH + lane + 64] = (_Float16)(a[lane + 64] * d2 * inv + bv[lane + 64]);
    dst[row * LDH + lane + 96] = (_Float16)(a[lane + 96] * d3 * inv + bv[lane + 96]);
  }
}

// ---------------------------------------------------------------- main kernel
__global__ __launch_bounds__(TPB, 1) void graformer_main(
    const float* __restrict__ x, const unsigned char* __restrict__ mask,
    const float* __restrict__ adj, const float* __restrict__ W_in,
    const float* __restrict__ b_in, const float* __restrict__ W_out,
    const float* __restrict__ b_out, const float* __restrict__ ln1_a,
    const float* __restrict__ ln1_b, const float* __restrict__ ln2_a,
    const float* __restrict__ ln2_b, const float* __restrict__ bq,
    const float* __restrict__ bk, const float* __restrict__ bv,
    const float* __restrict__ bo, const float* __restrict__ Ahat,
    const float* __restrict__ bg1, const float* __restrict__ bg2,
    const float* __restrict__ bc1, const float* __restrict__ bc2,
    const _Float16* __restrict__ wsH, float* __restrict__ out) {
  __shared__ SMem sm;
  const int tid = threadIdx.x, wid = tid >> 5, lane = tid & 31;
  const int b0 = blockIdx.x * BT;
  const int m0 = wid * 16;
  const int nD = lane & 15;
  const int mrB = m0 + ((lane & 16) ? 8 : 0);

  // ---- Laplacian from adj (row sums) + mask
  if (tid < NPTS) {
    float s = 0.f;
    for (int j = 0; j < NPTS; ++j) s += adj[tid * NPTS + j];
    sm.dcoef[tid] = rsqrtf(s);
    sm.maskl[tid] = mask[tid];
  }
  __syncthreads();
  if (tid < NPTS * NPTS) {
    int i = tid / NPTS, j = tid - i * NPTS;
    sm.lm[tid] = (i == j ? 1.f : 0.f) - sm.dcoef[i] * adj[tid] * sm.dcoef[j];
  }

  // ---- input ChebConv (K=2 channels, VALU)
  for (int idx = tid; idx < ROWS * 2; idx += TPB) {
    int row = idx >> 1, c = idx & 1;
    int bg = b0 + row / NPTS;
    sm.xl[idx] = (bg < BATCH) ? x[(bg * NPTS + row % NPTS) * 2 + c] : 0.f;
  }
  __syncthreads();
  for (int idx = tid; idx < ROWS * 2; idx += TPB) {
    int row = idx >> 1, c = idx & 1;
    int b = row / NPTS, n = row - b * NPTS, base = b * NPTS;
    float acc = 0.f;
    for (int j = 0; j < NPTS; ++j)
      acc += sm.lm[n * NPTS + j] * sm.xl[(base + j) * 2 + c];
    sm.t1c[idx] = acc;
  }
  __syncthreads();
  for (int idx = tid; idx < ROWS * 2; idx += TPB) {
    int row = idx >> 1, c = idx & 1;
    int b = row / NPTS, n = row - b * NPTS, base = b * NPTS;
    float acc = 0.f;
    for (int j = 0; j < NPTS; ++j)
      acc += sm.lm[n * NPTS + j] * sm.t1c[(base + j) * 2 + c];
    sm.t2c[idx] = 2.f * acc - sm.xl[idx];
  }
  __syncthreads();
  for (int idx = tid; idx < RP * HID; idx += TPB) {
    int row = idx >> 7, d = idx & 127;
    float acc = 0.f;
    if (row < ROWS) {
      acc = b_in[d];
      for (int c = 0; c < 2; ++c) {
        acc += sm.xl[row * 2 + c]  * W_in[c * 128 + d];
        acc += sm.t1c[row * 2 + c] * W_in[256 + c * 128 + d];
        acc += sm.t2c[row * 2 + c] * W_in[512 + c * 128 + d];
      }
    }
    sm.actF[idx] = acc;
  }
  __syncthreads();

  // ---- encoder layers
  for (int l = 0; l < NLAYERS; ++l) {
    const _Float16* WL = wsH + l * LAYER_HALVES;
    if (l + 1 < NLAYERS)  // stream next layer's weights toward L2
      __builtin_prefetch(wsH + (l + 1) * LAYER_HALVES + tid * 896, 0, 0);

    // ======== attention block (pre-norm residual) ========
    layer_norm(sm.actF, ln1_a + l * HID, ln1_b + l * HID, sm.sA, wid, lane);
    __syncthreads();

    // Q -> sB[:,0:128], K -> sB[:,128:256], V -> sA[:,128:256]
    for (int nt = 0; nt < 8; ++nt) {
      int n0 = nt * 16;
      v8f c = {0.f, 0.f, 0.f, 0.f, 0.f, 0.f, 0.f, 0.f};
      c = mm_part(c, sm.sA, LDH, m0, WL + OFF_WQ, HID, n0, lane);
      int n = n0 + nD;
#pragma unroll
      for (int r = 0; r < 8; ++r)
        sm.sB[(mrB + r) * LDH + n] = (_Float16)(c[r] + bq[l * HID + n]);
    }
    for (int nt = 0; nt < 8; ++nt) {
      int n0 = nt * 16;
      v8f c = {0.f, 0.f, 0.f, 0.f, 0.f, 0.f, 0.f, 0.f};
      c = mm_part(c, sm.sA, LDH, m0, WL + OFF_WK, HID, n0, lane);
      int n = n0 + nD;
#pragma unroll
      for (int r = 0; r < 8; ++r)
        sm.sB[(mrB + r) * LDH + 128 + n] = (_Float16)(c[r] + bk[l * HID + n]);
    }
    for (int nt = 0; nt < 8; ++nt) {
      int n0 = nt * 16;
      v8f c = {0.f, 0.f, 0.f, 0.f, 0.f, 0.f, 0.f, 0.f};
      c = mm_part(c, sm.sA, LDH, m0, WL + OFF_WV, HID, n0, lane);
      int n = n0 + nD;
#pragma unroll
      for (int r = 0; r < 8; ++r)
        sm.sA[(mrB + r) * LDH + 128 + n] = (_Float16)(c[r] + bv[l * HID + n]);
    }
    __syncthreads();

    // scores + softmax (21x21 per head, VALU, vectorized dot)
    float pb[2][NPTS];
    for (int it = 0; it < 2; ++it) {
      int item = tid + it * TPB;
      if (item < BT * NHEAD * NPTS) {
        int bl = item / (NHEAD * NPTS);
        int rem = item - bl * NHEAD * NPTS;
        int hd = rem / NPTS, q = rem - hd * NPTS;
        const _Float16* Qp = sm.sB + (bl * NPTS + q) * LDH + hd * 32;
        float sc[NPTS], mx = -1e30f;
        for (int k = 0; k < NPTS; ++k) {
          const _Float16* Kp = sm.sB + (bl * NPTS + k) * LDH + 128 + hd * 32;
          float s = 0.f;
#pragma unroll
          for (int d0 = 0; d0 < 32; d0 += 8) {
            v8h qv = *(const v8h*)(Qp + d0);
            v8h kv = *(const v8h*)(Kp + d0);
#pragma unroll
            for (int i = 0; i < 8; ++i) s += (float)qv[i] * (float)kv[i];
          }
          s *= 0.17677669529663687f;   // 1/sqrt(32)
          if (!sm.maskl[k]) s = -1e9f;
          sc[k] = s;
          mx = fmaxf(mx, s);
        }
        float sum = 0.f;
        for (int k = 0; k < NPTS; ++k) {
          float e = __expf(sc[k] - mx);
          pb[it][k] = e;
          sum += e;
        }
        float inv = 1.f / sum;
        for (int k = 0; k < NPTS; ++k) pb[it][k] *= inv;
      }
    }
    __syncthreads();
    // P@V, vectorized over head-dim
    for (int it = 0; it < 2; ++it) {
      int item = tid + it * TPB;
      if (item < BT * NHEAD * NPTS) {
        int bl = item / (NHEAD * NPTS);
        int rem = item - bl * NHEAD * NPTS;
        int hd = rem / NPTS, q = rem - hd * NPTS;
#pragma unroll
        for (int d0 = 0; d0 < 32; d0 += 8) {
          float o[8] = {0.f, 0.f, 0.f, 0.f, 0.f, 0.f, 0.f, 0.f};
          for (int k = 0; k < NPTS; ++k) {
            float p = pb[it][k];
            v8h vv = *(const v8h*)(sm.sA + (bl * NPTS + k) * LDH + 128 +
                                   hd * 32 + d0);
#pragma unroll
            for (int i = 0; i < 8; ++i) o[i] += p * (float)vv[i];
          }
          v8h ov;
#pragma unroll
          for (int i = 0; i < 8; ++i) ov[i] = (_Float16)o[i];
          *(v8h*)(sm.sB + (bl * NPTS + q) * LDH + hd * 32 + d0) = ov;
        }
      }
    }
    __syncthreads();

    // output projection + residual add
    for (int nt = 0; nt < 8; ++nt) {
      int n0 = nt * 16;
      v8f c = {0.f, 0.f, 0.f, 0.f, 0.f, 0.f, 0.f, 0.f};
      c = mm_part(c, sm.sB, LDH, m0, WL + OFF_WO, HID, n0, lane);
      int n = n0 + nD;
#pragma unroll
      for (int r = 0; r < 8; ++r)
        sm.actF[(mrB + r) * HID + n] += c[r] + bo[l * HID + n];
    }
    __syncthreads();

    // ======== GraphNet block (pre-norm residual) ========
    layer_norm(sm.actF, ln2_a + l * HID, ln2_b + l * HID, sm.sA, wid, lane);
    if (tid < NPTS) {  // A_hat normalization (column sums)
      float s = 0.f;
      for (int i = 0; i < NPTS; ++i) s += Ahat[l * 441 + i * NPTS + tid];
      sm.dcoef[tid] = rsqrtf(s + 1e-5f);
    }
    __syncthreads();
    if (tid < NPTS * NPTS) {
      int i = tid / NPTS, j = tid - i * NPTS;
      sm.lg[tid] = sm.dcoef[i] * Ahat[l * 441 + tid] * sm.dcoef[j];
    }
    __syncthreads();
    mix_nodes(sm.lg, sm.sA, LDH, sm.sB, LDH, HID, (const _Float16*)nullptr, 0);
    __syncthreads();
    // G1: 128 -> 256, relu -> sA[:,0:256]
    for (int nt = 0; nt < 16; ++nt) {
      int n0 = nt * 16;
      v8f c = {0.f, 0.f, 0.f, 0.f, 0.f, 0.f, 0.f, 0.f};
      c = mm_part(c, sm.sB, LDH, m0, WL + OFF_WG1, HID, n0, lane);
      int n = n0 + nD;
#pragma unroll
      for (int r = 0; r < 8; ++r)
        sm.sA[(mrB + r) * LDH + n] = (_Float16)fmaxf(c[r] + bg1[l * 256 + n], 0.f);
    }
    __syncthreads();
    mix_nodes(sm.lg, sm.sA, LDH, sm.sB, LDH, 2 * HID, (const _Float16*)nullptr, 0);
    __syncthreads();
    // G2: 256 -> 128, residual add
    for (int nt = 0; nt < 8; ++nt) {
      int n0 = nt * 16;
      v8f c = {0.f, 0.f, 0.f, 0.f, 0.f, 0.f, 0.f, 0.f};
      c = mm_part(c, sm.sB, LDH, m0, WL + OFF_WG2, 2 * HID, n0, lane);
      int n = n0 + nD;
#pragma unroll
      for (int r = 0; r < 8; ++r)
        sm.actF[(mrB + r) * HID + n] += c[r] + bg2[l * HID + n];
    }
    __syncthreads();

    // ======== ResChebGC: out += relu(cheb2(relu(cheb1(out)))) ========
    for (int idx = tid; idx < ROWS * (HID / 2); idx += TPB) {
      int row = idx >> 6, c = (idx & 63) * 2;
      h2 o;
      o[0] = (_Float16)sm.actF[row * HID + c];
      o[1] = (_Float16)sm.actF[row * HID + c + 1];
      *(h2*)(sm.sA + row * LDH + c) = o;
    }
    __syncthreads();
    mix_nodes(sm.lm, sm.sA, LDH, sm.sB, LDH, HID, (const _Float16*)nullptr, 0);
    __syncthreads();
    mix_nodes(sm.lm, sm.sB, LDH, sm.sB + 128, LDH, HID, sm.sA, LDH);  // t2
    __syncthreads();
    for (int nt = 0; nt < 8; ++nt) {  // cheb1 -> g = relu(.) in sA[:,128:256]
      int n0 = nt * 16;
      v8f c = {0.f, 0.f, 0.f, 0.f, 0.f, 0.f, 0.f, 0.f};
      c = mm_part(c, sm.sA, LDH, m0, WL + OFF_WC1, HID, n0, lane);
      c = mm_part(c, sm.sB, LDH, m0, WL + OFF_WC1 + 16384, HID, n0, lane);
      c = mm_part(c, sm.sB + 128, LDH, m0, WL + OFF_WC1 + 32768, HID, n0, lane);
      int n = n0 + nD;
#pragma unroll
      for (int r = 0; r < 8; ++r)
        sm.sA[(mrB + r) * LDH + 128 + n] =
            (_Float16)fmaxf(c[r] + bc1[l * HID + n], 0.f);
    }
    __syncthreads();
    mix_nodes(sm.lm, sm.sA + 128, LDH, sm.sB, LDH, HID, (const _Float16*)nullptr, 0);
    __syncthreads();
    mix_nodes(sm.lm, sm.sB, LDH, sm.sB + 128, LDH, HID, sm.sA + 128, LDH);
    __syncthreads();
    for (int nt = 0; nt < 8; ++nt) {  // cheb2, relu, residual add
      int n0 = nt * 16;
      v8f c = {0.f, 0.f, 0.f, 0.f, 0.f, 0.f, 0.f, 0.f};
      c = mm_part(c, sm.sA + 128, LDH, m0, WL + OFF_WC2, HID, n0, lane);
      c = mm_part(c, sm.sB, LDH, m0, WL + OFF_WC2 + 16384, HID, n0, lane);
      c = mm_part(c, sm.sB + 128, LDH, m0, WL + OFF_WC2 + 32768, HID, n0, lane);
      int n = n0 + nD;
#pragma unroll
      for (int r = 0; r < 8; ++r)
        sm.actF[(mrB + r) * HID + n] += fmaxf(c[r] + bc2[l * HID + n], 0.f);
    }
    __syncthreads();
  }

  // ---- output ChebConv (128 -> 3, VALU)
  for (int idx = tid; idx < ROWS * (HID / 2); idx += TPB) {
    int row = idx >> 6, c = (idx & 63) * 2;
    h2 o;
    o[0] = (_Float16)sm.actF[row * HID + c];
    o[1] = (_Float16)sm.actF[row * HID + c + 1];
    *(h2*)(sm.sA + row * LDH + c) = o;
  }
  __syncthreads();
  mix_nodes(sm.lm, sm.sA, LDH, sm.sB, LDH, HID, (const _Float16*)nullptr, 0);
  __syncthreads();
  mix_nodes(sm.lm, sm.sB, LDH, sm.sB + 128, LDH, HID, sm.sA, LDH);
  __syncthreads();
  for (int idx = tid; idx < ROWS * 3; idx += TPB) {
    int row = idx / 3, d = idx - row * 3;
    int bg = b0 + row / NPTS;
    if (bg < BATCH) {
      float acc = b_out[d];
      for (int c = 0; c < HID; ++c) {
        acc += (float)sm.sA[row * LDH + c] * W_out[c * 3 + d];
        acc += (float)sm.sB[row * LDH + c] * W_out[384 + c * 3 + d];
        acc += (float)sm.sB[row * LDH + 128 + c] * W_out[768 + c * 3 + d];
      }
      out[(bg * NPTS + row % NPTS) * 3 + d] = acc;
    }
  }
}

// ---------------------------------------------------------------- launch
extern "C" void kernel_launch(void* const* d_in, const int* in_sizes, int n_in,
                              void* d_out, int out_size, void* d_ws,
                              size_t ws_size, hipStream_t stream) {
  const float* x     = (const float*)d_in[0];
  const unsigned char* mask = (const unsigned char*)d_in[1];
  const float* adj   = (const float*)d_in[2];
  const float* W_in  = (const float*)d_in[3];
  const float* b_in  = (const float*)d_in[4];
  const float* W_out = (const float*)d_in[5];
  const float* b_out = (const float*)d_in[6];
  const float* ln1_a = (const float*)d_in[7];
  const float* ln1_b = (const float*)d_in[8];
  const float* ln2_a = (const float*)d_in[9];
  const float* ln2_b = (const float*)d_in[10];
  const float* Wq = (const float*)d_in[11];
  const float* bq = (const float*)d_in[12];
  const float* Wk = (const float*)d_in[13];
  const float* bk = (const float*)d_in[14];
  const float* Wv = (const float*)d_in[15];
  const float* bv = (const float*)d_in[16];
  const float* Wo = (const float*)d_in[17];
  const float* bo = (const float*)d_in[18];
  const float* Ahat = (const float*)d_in[19];
  const float* Wg1 = (const float*)d_in[20];
  const float* bg1 = (const float*)d_in[21];
  const float* Wg2 = (const float*)d_in[22];
  const float* bg2 = (const float*)d_in[23];
  const float* Wc1 = (const float*)d_in[24];
  const float* bc1 = (const float*)d_in[25];
  const float* Wc2 = (const float*)d_in[26];
  const float* bc2 = (const float*)d_in[27];

  _Float16* wsH = (_Float16*)d_ws;

  graformer_prep<<<TOTAL_HALVES / TPB, TPB, 0, stream>>>(Wq, Wk, Wv, Wo, Wg1,
                                                         Wg2, Wc1, Wc2, wsH);
  int nblocks = (BATCH + BT - 1) / BT;
  graformer_main<<<nblocks, TPB, 0, stream>>>(
      x, mask, adj, W_in, b_in, W_out, b_out, ln1_a, ln1_b, ln2_a, ln2_b, bq,
      bk, bv, bo, Ahat, bg1, bg2, bc1, bc2, wsH, (float*)d_out);
}
